// MASDE_56805237457263
// MI455X (gfx1250) — matrix operand order repaired and checked by reading, not compiled, not run on hardware
//
#include <hip/hip_runtime.h>
#include <hip/hip_bf16.h>
#include <math.h>

typedef __attribute__((ext_vector_type(16))) _Float16 v16h;
typedef __attribute__((ext_vector_type(8)))  float    v8f;

// ---------------- WMMA helpers (CDNA5 16x16x32 f16 -> f32) ----------------

__device__ __forceinline__ v8f wmma_(v16h a, v16h b, v8f c) {
  // 8 args: (neg_a, A, neg_b, B, c_mod, C, reuse_a, reuse_b)
  return __builtin_amdgcn_wmma_f32_16x16x32_f16(false, a, false, b, (short)0, c,
                                                false, false);
}

__device__ __forceinline__ int lane_() { return threadIdx.x & 31; }
__device__ __forceinline__ int cn_()   { return threadIdx.x & 15; }
__device__ __forceinline__ int cm_(int r) { return (((threadIdx.x & 31) >> 4) << 3) + r; }

// A fragment (16x32, f16) from row-major f32 source, leading dim ld.
__device__ __forceinline__ v16h frag_a_f32(const float* __restrict__ s, int ld) {
  const int lane = lane_();
  const int m = lane & 15;
  const int kb = (lane >> 4) << 3;          // 0 or 8
  const float* r = s + m * ld;
  v16h a;
#pragma unroll
  for (int i = 0; i < 8; ++i) {
    a[i]     = (_Float16)r[kb + i];         // K = kb..kb+7
    a[8 + i] = (_Float16)r[16 + kb + i];    // K = 16+kb..
  }
  return a;
}

// A fragment (16x32) from f16 source (LDS or global), leading dim ld.
__device__ __forceinline__ v16h frag_a_h(const _Float16* s, int ld) {
  const int lane = lane_();
  const int m = lane & 15;
  const int kb = (lane >> 4) << 3;
  const _Float16* r = s + m * ld;
  v16h a;
#pragma unroll
  for (int i = 0; i < 8; ++i) {
    a[i]     = r[kb + i];
    a[8 + i] = r[16 + kb + i];
  }
  return a;
}

// B fragment from pre-packed fragment-major f16 weights: each tile is
// 32 lanes x 16 halfs contiguous; lane chunk is one 32-byte aligned run.
__device__ __forceinline__ v16h frag_b_pk(const _Float16* __restrict__ pk) {
  const _Float16* r = pk + (size_t)lane_() * 16;
  v16h b;
#pragma unroll
  for (int i = 0; i < 16; ++i) b[i] = r[i];
  return b;
}

// ---------------- scalar math helpers ----------------

__device__ __forceinline__ float sigf(float x) { return 1.f / (1.f + expf(-x)); }
__device__ __forceinline__ float softplusf(float x) {
  return (x > 20.f) ? x : log1pf(expf(x));
}

// ---------------- model dims ----------------
#define BB 128
#define TT 192
#define FF 64
#define ZZ 64

// ============================================================
// zero accumulators
// ============================================================
__global__ __launch_bounds__(64) void k_zero(float* acc) { acc[threadIdx.x] = 0.f; }

// ============================================================
// Pack a (numNT*16, K) row-major f32 weight (leading dim ld) into
// fragment-major f16: pk[(nt*numKS+ks)*512 + lane*16 + i]
//   = W[(nt*16 + (lane&15))*ld + ks*32 + ((lane>>4)<<3) + (i<8 ? i : 8+i)]
// ============================================================
__global__ __launch_bounds__(256) void k_pack(const float* __restrict__ W,
                                              _Float16* __restrict__ pk,
                                              int numNT, int numKS, int ld) {
  const int u = blockIdx.x * blockDim.x + threadIdx.x;  // one per (tile,lane)
  const int total = numNT * numKS * 32;
  if (u >= total) return;
  const int lane = u & 31;
  const int tile = u >> 5;
  const int ks = tile % numKS;
  const int nt = tile / numKS;
  const float* s = W + (size_t)(nt * 16 + (lane & 15)) * ld + ks * 32 +
                   ((lane >> 4) << 3);
  _Float16* d = pk + (size_t)u * 16;
#pragma unroll
  for (int i = 0; i < 8; ++i) { d[i] = (_Float16)s[i]; d[8 + i] = (_Float16)s[16 + i]; }
}

// ============================================================
// gi = inp @ Wih^T + bih   (24576 x 384 x 128), Wih pre-packed f16
// ============================================================
__global__ __launch_bounds__(256) void k_gi(const float* __restrict__ inp,
                                            const _Float16* __restrict__ pW,
                                            const float* __restrict__ bias,
                                            float* __restrict__ gi) {
  const int wv = threadIdx.x >> 5;
  const int mt = wv & 3, half = wv >> 2;
  const int r0 = blockIdx.x * 64 + mt * 16;
  const int n = cn_();
  v16h a[4];
#pragma unroll
  for (int ks = 0; ks < 4; ++ks)
    a[ks] = frag_a_f32(inp + (size_t)r0 * 128 + ks * 32, 128);
  for (int j = 0; j < 12; ++j) {
    const int nt = half * 12 + j;
    const int n0 = nt * 16;
    v8f c;
#pragma unroll
    for (int r = 0; r < 8; ++r) c[r] = bias[n0 + n];
#pragma unroll
    for (int ks = 0; ks < 4; ++ks)
      c = wmma_(a[ks], frag_b_pk(pW + (size_t)(nt * 4 + ks) * 512), c);
#pragma unroll
    for (int r = 0; r < 8; ++r)
      gi[(size_t)(r0 + cm_(r)) * 384 + n0 + n] = c[r];
  }
}

// ============================================================
// GRU scan (single persistent workgroup, h in LDS as f16)
// + encoder heads + z0 (f16) + KL
// ============================================================
__global__ __launch_bounds__(1024) void k_gru(
    const float* __restrict__ gi, const _Float16* __restrict__ pWhh,
    const float* __restrict__ bhh, const _Float16* __restrict__ pMu,
    const float* __restrict__ mub, const _Float16* __restrict__ pLv,
    const float* __restrict__ lvb, const float* __restrict__ eps0,
    _Float16* __restrict__ zph, float* __restrict__ acc) {
  __shared__ _Float16 hbuf[128 * 132];
  __shared__ float sred[32];
  const int tid = threadIdx.x, wv = tid >> 5;
  const int n = cn_();
  for (int i = tid; i < 128 * 132; i += 1024) hbuf[i] = (_Float16)0.f;
  __syncthreads();
  const int mt = wv >> 2, np = wv & 3;
  for (int t = 0; t < TT; ++t) {
    v16h a[4];
#pragma unroll
    for (int ks = 0; ks < 4; ++ks)
      a[ks] = frag_a_h(hbuf + mt * 16 * 132 + ks * 32, 132);
    __syncthreads();   // all A-fragment reads of h done before any h' write
    if (t + 1 < TT) {  // prefetch next step's gi row for this lane
      const size_t gpf = ((size_t)(mt * 16 + cm_(0)) * TT + (t + 1)) * 384 +
                         np * 32 + n;
      __builtin_prefetch(gi + gpf, 0, 3);
    }
#pragma unroll
    for (int q = 0; q < 2; ++q) {
      const int nt = np * 2 + q;
      const int j0 = nt * 16 + n;
      v8f cr, cz, cg;
#pragma unroll
      for (int r = 0; r < 8; ++r) {
        cr[r] = bhh[j0]; cz[r] = bhh[128 + j0]; cg[r] = bhh[256 + j0];
      }
#pragma unroll
      for (int ks = 0; ks < 4; ++ks) {
        // gate g occupies row-tiles g*8+nt of the 24-tile packed Whh
        v16h br = frag_b_pk(pWhh + (size_t)((0 * 8 + nt) * 4 + ks) * 512);
        v16h bz = frag_b_pk(pWhh + (size_t)((1 * 8 + nt) * 4 + ks) * 512);
        v16h bn = frag_b_pk(pWhh + (size_t)((2 * 8 + nt) * 4 + ks) * 512);
        cr = wmma_(a[ks], br, cr);
        cz = wmma_(a[ks], bz, cz);
        cg = wmma_(a[ks], bn, cg);
      }
#pragma unroll
      for (int r = 0; r < 8; ++r) {
        const int mrow = mt * 16 + cm_(r);   // batch row
        const size_t gb = ((size_t)mrow * TT + t) * 384 + j0;
        float ir = gi[gb], iz = gi[gb + 128], ig = gi[gb + 256];
        float rg = sigf(ir + cr[r]);
        float u  = sigf(iz + cz[r]);
        float nn = tanhf(ig + rg * cg[r]);
        float hold = (float)hbuf[mrow * 132 + j0];
        hbuf[mrow * 132 + j0] = (_Float16)((1.f - u) * nn + u * hold);
      }
    }
    __syncthreads();
  }
  // encoder heads: mu0 / logv0, z0, KL   (32 (mt,nt) tiles, one per wave)
  {
    const int emt = wv >> 2, ent = wv & 3;
    v16h a[4];
#pragma unroll
    for (int ks = 0; ks < 4; ++ks)
      a[ks] = frag_a_h(hbuf + emt * 16 * 132 + ks * 32, 132);
    v8f cmu, clv;
#pragma unroll
    for (int r = 0; r < 8; ++r) { cmu[r] = mub[ent * 16 + n]; clv[r] = lvb[ent * 16 + n]; }
#pragma unroll
    for (int ks = 0; ks < 4; ++ks) {
      cmu = wmma_(a[ks], frag_b_pk(pMu + (size_t)(ent * 4 + ks) * 512), cmu);
      clv = wmma_(a[ks], frag_b_pk(pLv + (size_t)(ent * 4 + ks) * 512), clv);
    }
    float kls = 0.f;
#pragma unroll
    for (int r = 0; r < 8; ++r) {
      const int b = emt * 16 + cm_(r), z = ent * 16 + n;
      float mu = cmu[r], lv = clv[r];
      float z0 = mu + expf(0.5f * lv) * eps0[b * ZZ + z];
      zph[(size_t)b * ZZ + z] = (_Float16)z0;  // z_paths[0] (f16)
      kls += 0.5f * (expf(lv) + mu * mu - 1.f - lv);
    }
    for (int o = 16; o; o >>= 1) kls += __shfl_xor(kls, o, 32);
    if ((tid & 31) == 0) sred[wv] = kls;
    __syncthreads();
    if (wv == 0) {
      float v = sred[tid & 31];
      for (int o = 16; o; o >>= 1) v += __shfl_xor(v, o, 32);
      if (tid == 0) acc[3] = v / (float)BB;   // KL (BETA=1)
    }
  }
}

// ============================================================
// SDE Euler-Maruyama scan: 8 blocks x 16 batch rows, all linears via WMMA,
// all weights pre-packed f16
// ============================================================
__global__ __launch_bounds__(256) void k_sde(
    _Float16* __restrict__ zph, const float* __restrict__ dWn,
    const _Float16* pFW1, const float* fb1, const _Float16* pFW2, const float* fb2,
    const _Float16* pFW3, const float* fb3, const _Float16* pGW1, const float* gb1,
    const _Float16* pGW2, const float* gb2, const _Float16* pPW1, const float* pb1,
    const _Float16* pPW2, const float* pb2, const _Float16* pPeW, const float* peb,
    const _Float16* pWfp, const _Float16* pWgp) {
  __shared__ _Float16 zh[16 * 72];
  __shared__ _Float16 h1[16 * 136];
  __shared__ _Float16 h2g[16 * 136];
  __shared__ _Float16 h3p[16 * 136];
  __shared__ _Float16 f0b[16 * 72];
  __shared__ _Float16 g0b[16 * 72];
  __shared__ _Float16 spb[16 * 72];
  __shared__ float tmpb[16 * 68];
  const int tid = threadIdx.x, wv = tid >> 5;
  const int m0 = blockIdx.x * 16;
  const int n = cn_();
  const float dt = 1.f / 48.f;
  const float sqdt = sqrtf(dt);
  for (int i = tid; i < 16 * 64; i += 256) {
    int r = i >> 6, c = i & 63;
    zh[r * 72 + c] = zph[(size_t)(m0 + r) * ZZ + c];
  }
  __syncthreads();
  for (int t = 0; t < TT - 1; ++t) {
    // ---- phase 1: three Z->H linears (f0 h1 / g0 h / pih h)
    v16h az[2];
    az[0] = frag_a_h(zh, 72);
    az[1] = frag_a_h(zh + 32, 72);
#pragma unroll
    for (int jj = 0; jj < 3; ++jj) {
      const int job = wv + jj * 8;
      const int gm = job >> 3, nt = job & 7;
      const _Float16* pW = (gm == 0) ? pFW1 : (gm == 1) ? pGW1 : pPW1;
      const float* bb = (gm == 0) ? fb1 : (gm == 1) ? gb1 : pb1;
      _Float16* dst = (gm == 0) ? h1 : (gm == 1) ? h2g : h3p;
      v8f c;
#pragma unroll
      for (int r = 0; r < 8; ++r) c[r] = bb[nt * 16 + n];
      c = wmma_(az[0], frag_b_pk(pW + (size_t)(nt * 2 + 0) * 512), c);
      c = wmma_(az[1], frag_b_pk(pW + (size_t)(nt * 2 + 1) * 512), c);
#pragma unroll
      for (int r = 0; r < 8; ++r) {
        float v = c[r];
        float o = (gm == 2) ? v * sigf(v) : tanhf(v);   // silu for pih, tanh else
        dst[cm_(r) * 136 + nt * 16 + n] = (_Float16)o;
      }
    }
    __syncthreads();
    // ---- phase 2: h2 = tanh(h1@fW2+b) in-place ; g0 = h2g@gW2+b ; pi = sigmoid(h3p@pW2+b)
    {
      v16h fa[4], fb_[4];
#pragma unroll
      for (int ks = 0; ks < 4; ++ks) fa[ks] = frag_a_h(h1 + ks * 32, 136);
      const _Float16* src2 = (wv < 4) ? h2g : h3p;
#pragma unroll
      for (int ks = 0; ks < 4; ++ks) fb_[ks] = frag_a_h(src2 + ks * 32, 136);
      __syncthreads();   // all reads of h1 done before in-place write
      {
        const int nt = wv;
        v8f c;
#pragma unroll
        for (int r = 0; r < 8; ++r) c[r] = fb2[nt * 16 + n];
#pragma unroll
        for (int ks = 0; ks < 4; ++ks)
          c = wmma_(fa[ks], frag_b_pk(pFW2 + (size_t)(nt * 4 + ks) * 512), c);
#pragma unroll
        for (int r = 0; r < 8; ++r) h1[cm_(r) * 136 + nt * 16 + n] = (_Float16)tanhf(c[r]);
      }
      {
        const int nt = (wv < 4) ? wv : wv - 4;
        const _Float16* pW = (wv < 4) ? pGW2 : pPW2;
        const float* bb = (wv < 4) ? gb2 : pb2;
        v8f c;
#pragma unroll
        for (int r = 0; r < 8; ++r) c[r] = bb[nt * 16 + n];
#pragma unroll
        for (int ks = 0; ks < 4; ++ks)
          c = wmma_(fb_[ks], frag_b_pk(pW + (size_t)(nt * 4 + ks) * 512), c);
        if (wv < 4) {
#pragma unroll
          for (int r = 0; r < 8; ++r) g0b[cm_(r) * 72 + nt * 16 + n] = (_Float16)c[r];
        } else {
#pragma unroll
          for (int r = 0; r < 8; ++r) spb[cm_(r) * 72 + nt * 16 + n] = (_Float16)sigf(c[r]);
        }
      }
    }
    __syncthreads();
    // ---- phase 3: f0 = h2@fW3+b (waves 0-3) ; s_pi = tanh(pi@peW+b) in-place (waves 4-7)
    {
      v16h fa4[4]; v16h sa[2];
      if (wv < 4) {
#pragma unroll
        for (int ks = 0; ks < 4; ++ks) fa4[ks] = frag_a_h(h1 + ks * 32, 136);
      } else {
        sa[0] = frag_a_h(spb, 72);
        sa[1] = frag_a_h(spb + 32, 72);
      }
      __syncthreads();
      if (wv < 4) {
        const int nt = wv;
        v8f c;
#pragma unroll
        for (int r = 0; r < 8; ++r) c[r] = fb3[nt * 16 + n];
#pragma unroll
        for (int ks = 0; ks < 4; ++ks)
          c = wmma_(fa4[ks], frag_b_pk(pFW3 + (size_t)(nt * 4 + ks) * 512), c);
#pragma unroll
        for (int r = 0; r < 8; ++r) f0b[cm_(r) * 72 + nt * 16 + n] = (_Float16)c[r];
      } else {
        const int nt = wv - 4;
        v8f c;
#pragma unroll
        for (int r = 0; r < 8; ++r) c[r] = peb[nt * 16 + n];
        c = wmma_(sa[0], frag_b_pk(pPeW + (size_t)(nt * 2 + 0) * 512), c);
        c = wmma_(sa[1], frag_b_pk(pPeW + (size_t)(nt * 2 + 1) * 512), c);
#pragma unroll
        for (int r = 0; r < 8; ++r) spb[cm_(r) * 72 + nt * 16 + n] = (_Float16)tanhf(c[r]);
      }
    }
    __syncthreads();
    // ---- phase 4: f = f0 + s_pi@Wf ; g = softplus(g0 + s_pi@Wg) ; z update
    {
      v16h sa[2];
      sa[0] = frag_a_h(spb, 72);
      sa[1] = frag_a_h(spb + 32, 72);
      const int nt = (wv < 4) ? wv : wv - 4;
      const _Float16* cb = (wv < 4) ? f0b : g0b;
      v8f c;
#pragma unroll
      for (int r = 0; r < 8; ++r) c[r] = (float)cb[cm_(r) * 72 + nt * 16 + n];
      const _Float16* pW = (wv < 4) ? pWfp : pWgp;
      c = wmma_(sa[0], frag_b_pk(pW + (size_t)(nt * 2 + 0) * 512), c);
      c = wmma_(sa[1], frag_b_pk(pW + (size_t)(nt * 2 + 1) * 512), c);
      if (wv >= 4) {
#pragma unroll
        for (int r = 0; r < 8; ++r) {
          const int i = cm_(r), col = nt * 16 + n;
          float g = softplusf(c[r]);
          float dw = dWn[((size_t)t * BB + (m0 + i)) * ZZ + col];
          tmpb[i * 68 + col] = g * sqdt * dw;
        }
      }
      __syncthreads();
      if (wv < 4) {
#pragma unroll
        for (int r = 0; r < 8; ++r) {
          const int i = cm_(r), col = nt * 16 + n;
          float zo = (float)zh[i * 72 + col];
          float zn = zo + c[r] * dt + tmpb[i * 68 + col];
          zh[i * 72 + col] = (_Float16)zn;
          zph[((size_t)(t + 1) * BB + (m0 + i)) * ZZ + col] = (_Float16)zn;
        }
      }
    }
    __syncthreads();
  }
}

// ============================================================
// Decoder (two 64->256->64 MLPs) + Gaussian NLL partial sums
// ============================================================
__global__ __launch_bounds__(256) void k_dec(
    const _Float16* __restrict__ zph, const float* __restrict__ X,
    const float* __restrict__ O, const _Float16* pMW1, const float* mb1,
    const _Float16* pMW2, const float* mb2, const _Float16* pSW1,
    const float* sb1, const _Float16* pSW2, const float* sb2,
    float* __restrict__ acc) {
  __shared__ _Float16 hb[64 * 264];
  const int tid = threadIdx.x, wv = tid >> 5;
  const int r0 = blockIdx.x * 64;
  const int n = cn_();
  const int mtA = wv >> 1;
  v16h az[2];
  az[0] = frag_a_h(zph + (size_t)(r0 + mtA * 16) * 64, 64);
  az[1] = frag_a_h(zph + (size_t)(r0 + mtA * 16) * 64 + 32, 64);
  // phase 1: h = silu(z @ dmu_W1^T + b1)
#pragma unroll
  for (int i = 0; i < 8; ++i) {
    const int nt = (wv & 1) * 8 + i;
    v8f c;
#pragma unroll
    for (int r = 0; r < 8; ++r) c[r] = mb1[nt * 16 + n];
    c = wmma_(az[0], frag_b_pk(pMW1 + (size_t)(nt * 2 + 0) * 512), c);
    c = wmma_(az[1], frag_b_pk(pMW1 + (size_t)(nt * 2 + 1) * 512), c);
#pragma unroll
    for (int r = 0; r < 8; ++r) {
      float v = c[r];
      hb[(mtA * 16 + cm_(r)) * 264 + nt * 16 + n] = (_Float16)(v * sigf(v));
    }
  }
  __syncthreads();
  // phase 2: mu = h @ dmu_W2^T + b2   (kept in registers)
  v8f cmu[2];
  {
    v16h fa[8];
#pragma unroll
    for (int ks = 0; ks < 8; ++ks) fa[ks] = frag_a_h(hb + (mtA * 16) * 264 + ks * 32, 264);
#pragma unroll
    for (int q = 0; q < 2; ++q) {
      const int nt = (wv & 1) * 2 + q;
      v8f c;
#pragma unroll
      for (int r = 0; r < 8; ++r) c[r] = mb2[nt * 16 + n];
#pragma unroll
      for (int ks = 0; ks < 8; ++ks)
        c = wmma_(fa[ks], frag_b_pk(pMW2 + (size_t)(nt * 8 + ks) * 512), c);
      cmu[q] = c;
    }
  }
  __syncthreads();
  // phase 3: h = silu(z @ dls_W1^T + b1)  (overwrite hb)
#pragma unroll
  for (int i = 0; i < 8; ++i) {
    const int nt = (wv & 1) * 8 + i;
    v8f c;
#pragma unroll
    for (int r = 0; r < 8; ++r) c[r] = sb1[nt * 16 + n];
    c = wmma_(az[0], frag_b_pk(pSW1 + (size_t)(nt * 2 + 0) * 512), c);
    c = wmma_(az[1], frag_b_pk(pSW1 + (size_t)(nt * 2 + 1) * 512), c);
#pragma unroll
    for (int r = 0; r < 8; ++r) {
      float v = c[r];
      hb[(mtA * 16 + cm_(r)) * 264 + nt * 16 + n] = (_Float16)(v * sigf(v));
    }
  }
  __syncthreads();
  // phase 4: log_sig + elementwise NLL
  float lls = 0.f, obss = 0.f;
  {
    v16h fa[8];
#pragma unroll
    for (int ks = 0; ks < 8; ++ks) fa[ks] = frag_a_h(hb + (mtA * 16) * 264 + ks * 32, 264);
#pragma unroll
    for (int q = 0; q < 2; ++q) {
      const int nt = (wv & 1) * 2 + q;
      v8f c;
#pragma unroll
      for (int r = 0; r < 8; ++r) c[r] = sb2[nt * 16 + n];
#pragma unroll
      for (int ks = 0; ks < 8; ++ks)
        c = wmma_(fa[ks], frag_b_pk(pSW2 + (size_t)(nt * 8 + ks) * 512), c);
#pragma unroll
      for (int r = 0; r < 8; ++r) {
        const int row = r0 + mtA * 16 + cm_(r);
        const int t = row >> 7, b = row & 127;
        const int f = nt * 16 + n;
        const size_t xi = ((size_t)b * TT + t) * FF + f;
        float x = X[xi], o = O[xi];
        bool fin = (x - x == 0.f);            // finite check (NaN/Inf fail)
        float ob = (fin && o > 0.5f) ? 1.f : 0.f;
        float xt = fin ? x : 0.f;
        float ls = fminf(fmaxf(c[r], -7.f), 5.f);
        float s2 = fmaxf(expf(2.f * ls), 1e-6f);
        float d = xt - cmu[q][r];
        float ll = -0.5f * (1.8378770664093454f + logf(s2) + d * d / s2);
        lls += ll * ob;
        obss += ob;
      }
    }
  }
  for (int o = 16; o; o >>= 1) { lls += __shfl_xor(lls, o, 32); obss += __shfl_xor(obss, o, 32); }
  if ((tid & 31) == 0) {
    atomicAdd(acc + 0, lls);
    atomicAdd(acc + 1, obss);
  }
}

// ============================================================
// dt features (forward/backward time-since-observation)
// ============================================================
__global__ __launch_bounds__(256) void k_dt(const float* __restrict__ O,
                                            float* __restrict__ dfwd,
                                            float* __restrict__ dbwd) {
  const int q = blockIdx.x * blockDim.x + threadIdx.x;
  if (q >= BB * FF) return;
  const int b = q >> 6, f = q & 63;
  const float invT = 1.f / (float)TT;
  int last = -1;
  for (int t = 0; t < TT; ++t) {
    float o = O[((size_t)b * TT + t) * FF + f];
    if (o > 0.5f) last = t;
    dfwd[(size_t)q * TT + t] = (last >= 0 ? (float)(t - last) : (float)(t + 1)) * invT;
  }
  int nxt = -1;
  for (int t = TT - 1; t >= 0; --t) {
    float o = O[((size_t)b * TT + t) * FF + f];
    if (o > 0.5f) nxt = t;
    dbwd[(size_t)q * TT + t] = (nxt >= 0 ? (float)(nxt - t) : (float)(TT - t)) * invT;
  }
}

// ============================================================
// U = z_paths @ sel_W1[:, :64]^T + sel_b1  (z-part shared across F)
// ============================================================
__global__ __launch_bounds__(256) void k_selu(const _Float16* __restrict__ zph,
                                              const _Float16* __restrict__ pSel,
                                              const float* __restrict__ b1,
                                              float* __restrict__ U) {
  const int tid = threadIdx.x, wv = tid >> 5;
  const int r0 = blockIdx.x * 64;
  const int n = cn_();
  const int mt = wv >> 1;
  v16h a0 = frag_a_h(zph + (size_t)(r0 + mt * 16) * 64, 64);
  v16h a1 = frag_a_h(zph + (size_t)(r0 + mt * 16) * 64 + 32, 64);
#pragma unroll
  for (int q = 0; q < 2; ++q) {
    const int nt = (wv & 1) * 2 + q;
    v8f c;
#pragma unroll
    for (int r = 0; r < 8; ++r) c[r] = b1[nt * 16 + n];
    c = wmma_(a0, frag_b_pk(pSel + (size_t)(nt * 2 + 0) * 512), c);
    c = wmma_(a1, frag_b_pk(pSel + (size_t)(nt * 2 + 1) * 512), c);
#pragma unroll
    for (int r = 0; r < 8; ++r)
      U[(size_t)(r0 + mt * 16 + cm_(r)) * 64 + nt * 16 + n] = c[r];
  }
}

// ============================================================
// Selection head BCE: one wave per (b,f), 32 lanes cover SEL_H=64
// ============================================================
__global__ __launch_bounds__(256) void k_sel(const float* __restrict__ U,
                                             const float* __restrict__ dfwd,
                                             const float* __restrict__ dbwd,
                                             const float* __restrict__ O,
                                             const float* __restrict__ W1,
                                             const float* __restrict__ W2,
                                             const float* __restrict__ b2,
                                             float* __restrict__ acc) {
  const int tid = threadIdx.x;
  const int g = blockIdx.x * 8 + (tid >> 5);
  const int lane = tid & 31;
  const int b = g >> 6, f = g & 63;
  const float w1a0 = W1[lane * 66 + 64],        w1b0 = W1[lane * 66 + 65];
  const float w1a1 = W1[(lane + 32) * 66 + 64], w1b1 = W1[(lane + 32) * 66 + 65];
  const float w20 = W2[lane], w21 = W2[lane + 32];
  const float bias2 = b2[0];
  float bces = 0.f;
  for (int t = 0; t < TT; ++t) {
    const size_t urow = ((size_t)t * BB + b) * 64;
    const float u0 = U[urow + lane], u1 = U[urow + lane + 32];
    const size_t di = (size_t)(b * 64 + f) * TT + t;
    const float df = dfwd[di], db = dbwd[di];
    float p0 = u0 + df * w1a0 + db * w1b0;
    float p1 = u1 + df * w1a1 + db * w1b1;
    float s = p0 * sigf(p0) * w20 + p1 * sigf(p1) * w21;
    for (int o = 16; o; o >>= 1) s += __shfl_xor(s, o, 32);
    if (lane == 0) {
      float l = s + bias2;
      float R = 1.f - O[((size_t)b * TT + t) * FF + f];
      bces += fmaxf(l, 0.f) - l * R + log1pf(expf(-fabsf(l)));
    }
  }
  if (lane == 0) atomicAdd(acc + 2, bces);
}

// ============================================================
// Final combine
// ============================================================
__global__ __launch_bounds__(32) void k_final(const float* __restrict__ acc,
                                              float* __restrict__ out) {
  if (threadIdx.x == 0) {
    float nll = -acc[0] / fmaxf(acc[1], 1.f);
    float bce = acc[2] / (float)((size_t)BB * TT * FF);
    out[0] = nll + acc[3] + 0.2f * bce;
  }
}

// ============================================================
extern "C" void kernel_launch(void* const* d_in, const int* in_sizes, int n_in,
                              void* d_out, int out_size, void* d_ws, size_t ws_size,
                              hipStream_t stream) {
  (void)in_sizes; (void)n_in; (void)out_size; (void)ws_size;
  const float* inp     = (const float*)d_in[0];
  const float* O       = (const float*)d_in[1];
  const float* X       = (const float*)d_in[2];
  const float* eps0    = (const float*)d_in[3];
  const float* dWn     = (const float*)d_in[4];
  const float* gru_Wih = (const float*)d_in[5];
  const float* gru_bih = (const float*)d_in[6];
  const float* gru_Whh = (const float*)d_in[7];
  const float* gru_bhh = (const float*)d_in[8];
  const float* enc_muW = (const float*)d_in[9];
  const float* enc_mub = (const float*)d_in[10];
  const float* enc_lvW = (const float*)d_in[11];
  const float* enc_lvb = (const float*)d_in[12];
  const float* f0_W1 = (const float*)d_in[13]; const float* f0_b1 = (const float*)d_in[14];
  const float* f0_W2 = (const float*)d_in[15]; const float* f0_b2 = (const float*)d_in[16];
  const float* f0_W3 = (const float*)d_in[17]; const float* f0_b3 = (const float*)d_in[18];
  const float* g0_W1 = (const float*)d_in[19]; const float* g0_b1 = (const float*)d_in[20];
  const float* g0_W2 = (const float*)d_in[21]; const float* g0_b2 = (const float*)d_in[22];
  const float* pih_W1 = (const float*)d_in[23]; const float* pih_b1 = (const float*)d_in[24];
  const float* pih_W2 = (const float*)d_in[25]; const float* pih_b2 = (const float*)d_in[26];
  const float* pie_W = (const float*)d_in[27]; const float* pie_b = (const float*)d_in[28];
  const float* Wf = (const float*)d_in[29];    const float* Wg = (const float*)d_in[30];
  const float* dmu_W1 = (const float*)d_in[31]; const float* dmu_b1 = (const float*)d_in[32];
  const float* dmu_W2 = (const float*)d_in[33]; const float* dmu_b2 = (const float*)d_in[34];
  const float* dls_W1 = (const float*)d_in[35]; const float* dls_b1 = (const float*)d_in[36];
  const float* dls_W2 = (const float*)d_in[37]; const float* dls_b2 = (const float*)d_in[38];
  const float* sel_W1 = (const float*)d_in[39]; const float* sel_b1 = (const float*)d_in[40];
  const float* sel_W2 = (const float*)d_in[41]; const float* sel_b2 = (const float*)d_in[42];

  // ---- f32 workspace arenas ----
  float* ws  = (float*)d_ws;
  float* acc = ws;                                  // 64 floats of accumulators
  float* gi  = ws + 64;                             // 24576*384
  float* U   = gi + (size_t)24576 * 384;            // 24576*64
  float* dfw = U + (size_t)24576 * 64;              // 128*64*192
  float* dbw = dfw + (size_t)BB * FF * TT;          // 128*64*192
  // ---- f16 workspace arenas ----
  _Float16* zph  = (_Float16*)(dbw + (size_t)BB * FF * TT);  // z_paths (T,B,Z) f16
  _Float16* pWih = zph + (size_t)24576 * 64;
  _Float16* pWhh = pWih + 49152;
  _Float16* pMu  = pWhh + 49152;
  _Float16* pLv  = pMu + 8192;
  _Float16* pFW1 = pLv + 8192;
  _Float16* pFW2 = pFW1 + 8192;
  _Float16* pFW3 = pFW2 + 16384;
  _Float16* pGW1 = pFW3 + 8192;
  _Float16* pGW2 = pGW1 + 8192;
  _Float16* pPW1 = pGW2 + 8192;
  _Float16* pPW2 = pPW1 + 8192;
  _Float16* pPeW = pPW2 + 8192;
  _Float16* pWfp = pPeW + 4096;
  _Float16* pWgp = pWfp + 4096;
  _Float16* pMW1 = pWgp + 4096;
  _Float16* pMW2 = pMW1 + 16384;
  _Float16* pSW1 = pMW2 + 16384;
  _Float16* pSW2 = pSW1 + 16384;
  _Float16* pSel = pSW2 + 16384;

  auto packW = [&](const float* W, _Float16* dst, int numNT, int numKS, int ld) {
    const int total = numNT * numKS * 32;
    k_pack<<<(total + 255) / 256, 256, 0, stream>>>(W, dst, numNT, numKS, ld);
  };

  k_zero<<<1, 64, 0, stream>>>(acc);
  packW(gru_Wih, pWih, 24, 4, 128);
  packW(gru_Whh, pWhh, 24, 4, 128);
  packW(enc_muW, pMu, 4, 4, 128);
  packW(enc_lvW, pLv, 4, 4, 128);
  packW(f0_W1, pFW1, 8, 2, 64);
  packW(f0_W2, pFW2, 8, 4, 128);
  packW(f0_W3, pFW3, 4, 4, 128);
  packW(g0_W1, pGW1, 8, 2, 64);
  packW(g0_W2, pGW2, 4, 4, 128);
  packW(pih_W1, pPW1, 8, 2, 64);
  packW(pih_W2, pPW2, 4, 4, 128);
  packW(pie_W, pPeW, 4, 2, 64);
  packW(Wf, pWfp, 4, 2, 64);
  packW(Wg, pWgp, 4, 2, 64);
  packW(dmu_W1, pMW1, 16, 2, 64);
  packW(dmu_W2, pMW2, 4, 8, 256);
  packW(dls_W1, pSW1, 16, 2, 64);
  packW(dls_W2, pSW2, 4, 8, 256);
  packW(sel_W1, pSel, 4, 2, 66);

  k_gi<<<384, 256, 0, stream>>>(inp, pWih, gru_bih, gi);
  k_gru<<<1, 1024, 0, stream>>>(gi, pWhh, gru_bhh, pMu, enc_mub, pLv, enc_lvb,
                                eps0, zph, acc);
  k_sde<<<8, 256, 0, stream>>>(zph, dWn, pFW1, f0_b1, pFW2, f0_b2, pFW3, f0_b3,
                               pGW1, g0_b1, pGW2, g0_b2, pPW1, pih_b1, pPW2,
                               pih_b2, pPeW, pie_b, pWfp, pWgp);
  k_dec<<<384, 256, 0, stream>>>(zph, X, O, pMW1, dmu_b1, pMW2, dmu_b2, pSW1,
                                 dls_b1, pSW2, dls_b2, acc);
  k_dt<<<32, 256, 0, stream>>>(O, dfw, dbw);
  k_selu<<<384, 256, 0, stream>>>(zph, pSel, sel_b1, U);
  k_sel<<<1024, 256, 0, stream>>>(U, dfw, dbw, O, sel_W1, sel_W2, sel_b2, acc);
  k_final<<<1, 32, 0, stream>>>(acc, (float*)d_out);
}
